// EfficientVectorQuantiser_67972152427054
// MI455X (gfx1250) — matrix operand order repaired and checked by reading, not compiled, not run on hardware
//
#include <hip/hip_runtime.h>
#include <hip/hip_bf16.h>

typedef __attribute__((ext_vector_type(16))) _Float16 v16h;
typedef __attribute__((ext_vector_type(8)))  float    v8f;

#define N_VEC   16384   // 16*32*32 rows
#define K_CODE  8192    // codebook entries
#define DIM     64
#define HW      1024    // 32*32
#define CSTRIDE 1024    // stride between channels in z layout (b,c,h,w)
#define BSTRIDE 65536   // 64*1024, stride between batches
#define KCHUNK  2048    // codes per wave (4 chunks)
#define TILES_PER_CHUNK (KCHUNK / 16)   // 128

// ---------------- init: zero bincount + loss partials ----------------
__global__ void vq_init_kernel(int* __restrict__ bin_ws, float* __restrict__ partials) {
  int t = blockIdx.x * blockDim.x + threadIdx.x;
  if (t < K_CODE) bin_ws[t] = 0;
  if (t < 64)     partials[t] = 0.0f;
}

// ---------------- normalize codebook rows -> f16 ----------------
__global__ void vq_norm_cb_kernel(const float* __restrict__ embed, _Float16* __restrict__ cn) {
  int k = blockIdx.x * blockDim.x + threadIdx.x;
  if (k >= K_CODE) return;
  const float* row = embed + (size_t)k * DIM;
  float s = 0.0f;
  float vals[DIM];
#pragma unroll
  for (int c = 0; c < DIM; ++c) { float v = row[c]; vals[c] = v; s += v * v; }
  float inv = 1.0f / fmaxf(sqrtf(s), 1e-12f);
  _Float16* o = cn + (size_t)k * DIM;
#pragma unroll
  for (int c = 0; c < DIM; ++c) o[c] = (_Float16)(vals[c] * inv);
}

// ---------------- normalize z rows (b,c,h,w layout) -> f16 row-major ----------------
__global__ void vq_norm_z_kernel(const float* __restrict__ z, _Float16* __restrict__ zn) {
  int n = blockIdx.x * blockDim.x + threadIdx.x;
  if (n >= N_VEC) return;
  int b = n >> 10, hw = n & (HW - 1);
  const float* base = z + (size_t)b * BSTRIDE + hw;  // + c*CSTRIDE
  float s = 0.0f;
  float vals[DIM];
#pragma unroll
  for (int c = 0; c < DIM; ++c) { float v = base[(size_t)c * CSTRIDE]; vals[c] = v; s += v * v; }
  float inv = 1.0f / fmaxf(sqrtf(s), 1e-12f);
  _Float16* o = zn + (size_t)n * DIM;
#pragma unroll
  for (int c = 0; c < DIM; ++c) o[c] = (_Float16)(vals[c] * inv);
}

// ---------------- fused WMMA GEMM + per-row argmax ----------------
// Block = 256 threads = 8 waves. Wave w: row-tile rt = w&1 (local), K-chunk = w>>1.
// Each wave: 16 rows x 2048 codes, software-pipelined, 2 column tiles / iter
// with independent accumulators. Partial (max, argmax) combined via LDS.
__global__ void __launch_bounds__(256)
vq_argmax_kernel(const _Float16* __restrict__ zn, const _Float16* __restrict__ cn,
                 int* __restrict__ idx_out) {
  __shared__ float sval[8][16];
  __shared__ int   sidx[8][16];

  const int lane = threadIdx.x & 31;
  const int wave = threadIdx.x >> 5;
  const int rt    = wave & 1;                 // local row tile 0/1
  const int chunk = wave >> 1;                // 0..3
  const int rowBase = (blockIdx.x * 2 + rt) * 16;
  const int colBase = chunk * KCHUNK;
  const int g  = lane >> 4;                   // lane group (0/1)
  const int lm = lane & 15;                   // lane % 16

  // A fragment: lane lm holds row M=lm; half h -> K = h + g*16
  const _Float16* arow = zn + (size_t)(rowBase + lm) * DIM;
  v16h a0, a1;
#pragma unroll
  for (int h = 0; h < 16; ++h) {
    int k = h + g * 16;
    a0[h] = arow[k];
    a1[h] = arow[32 + k];
  }

  float best[8];
  int   bidx[8];
#pragma unroll
  for (int r = 0; r < 8; ++r) { best[r] = -3.4e38f; bidx[r] = 0; }

  // B fragment base: lane holds column n = colBase + t*16 + lm; halves -> K = h + g*16
  // one column tile = 16 rows * 64 halves = 1024 halves
  const _Float16* bptr = cn + (size_t)(colBase + lm) * DIM + g * 16;

  // prologue: preload tiles 0 and 1
  v16h b0a = *reinterpret_cast<const v16h*>(bptr);
  v16h b1a = *reinterpret_cast<const v16h*>(bptr + 32);
  v16h b0b = *reinterpret_cast<const v16h*>(bptr + 1024);
  v16h b1b = *reinterpret_cast<const v16h*>(bptr + 1056);
  bptr += 2048;

  for (int t = 0; t < TILES_PER_CHUNK; t += 2) {
    // preload tiles t+2, t+3 (last iteration reads a few KB past cn into
    // other mapped workspace regions; values unused)
    v16h n0a = *reinterpret_cast<const v16h*>(bptr);
    v16h n1a = *reinterpret_cast<const v16h*>(bptr + 32);
    v16h n0b = *reinterpret_cast<const v16h*>(bptr + 1024);
    v16h n1b = *reinterpret_cast<const v16h*>(bptr + 1056);
    __builtin_prefetch(bptr + 4096, 0, 1);    // warm tiles t+6.. -> global_prefetch_b8
    bptr += 2048;

    v8f accA = {};
    accA = __builtin_amdgcn_wmma_f32_16x16x32_f16(false, a0, false, b0a, (short)0, accA, false, false);
    accA = __builtin_amdgcn_wmma_f32_16x16x32_f16(false, a1, false, b1a, (short)0, accA, false, false);
    v8f accB = {};
    accB = __builtin_amdgcn_wmma_f32_16x16x32_f16(false, a0, false, b0b, (short)0, accB, false, false);
    accB = __builtin_amdgcn_wmma_f32_16x16x32_f16(false, a1, false, b1b, (short)0, accB, false, false);

    const int nA = colBase + t * 16 + lm;
    const int nB = nA + 16;
    // acc[r] = dist[row = g*8 + r][n]; strict > keeps first (lowest) index within lane
#pragma unroll
    for (int r = 0; r < 8; ++r) {
      float v = accA[r];
      if (v > best[r]) { best[r] = v; bidx[r] = nA; }
    }
#pragma unroll
    for (int r = 0; r < 8; ++r) {
      float v = accB[r];
      if (v > best[r]) { best[r] = v; bidx[r] = nB; }
    }

    b0a = n0a; b1a = n1a; b0b = n0b; b1b = n1b;
  }

  // Reduce across the 16 lanes (lm) that share the same 8 rows; tie-break lower index.
#pragma unroll
  for (int off = 1; off < 16; off <<= 1) {
#pragma unroll
    for (int r = 0; r < 8; ++r) {
      float ov = __shfl_xor(best[r], off);
      int   oi = __shfl_xor(bidx[r], off);
      if (ov > best[r] || (ov == best[r] && oi < bidx[r])) { best[r] = ov; bidx[r] = oi; }
    }
  }
  if (lm == 0) {
#pragma unroll
    for (int r = 0; r < 8; ++r) {
      sval[wave][g * 8 + r] = best[r];
      sidx[wave][g * 8 + r] = bidx[r];
    }
  }
  __syncthreads();

  // Waves 0/1 combine the 4 K-chunk partials for their row tile.
  if (wave < 2 && lane < 16) {
    const int m = lane;
    float bv = sval[rt][m];
    int   bi = sidx[rt][m];
#pragma unroll
    for (int c = 1; c < 4; ++c) {
      float v = sval[c * 2 + rt][m];
      int   i = sidx[c * 2 + rt][m];
      if (v > bv || (v == bv && i < bi)) { bv = v; bi = i; }
    }
    idx_out[rowBase + m] = bi;
  }
}

// ---------------- gather codebook rows, loss partials, bincount ----------------
__global__ void __launch_bounds__(256)
vq_gather_kernel(const float* __restrict__ z, const float* __restrict__ embed,
                 const int* __restrict__ idx_ws, float* __restrict__ out,
                 float* __restrict__ out_idx, int* __restrict__ bin_ws,
                 float* __restrict__ partials) {
  __shared__ float sred[256];
  int n = blockIdx.x * 256 + threadIdx.x;   // grid covers exactly N_VEC
  int idx = idx_ws[n];
  int b = n >> 10, hw = n & (HW - 1);
  const float* zb = z + (size_t)b * BSTRIDE + hw;
  const float* e  = embed + (size_t)idx * DIM;
  float* ob = out + (size_t)b * BSTRIDE + hw;
  float local = 0.0f;
#pragma unroll
  for (int c = 0; c < DIM; ++c) {
    float q = e[c];
    float d = q - zb[(size_t)c * CSTRIDE];
    local += d * d;
    ob[(size_t)c * CSTRIDE] = q;            // straight-through output == z_q
  }
  out_idx[n] = (float)idx;
  atomicAdd(&bin_ws[idx], 1);               // integer: exact, order-independent

  sred[threadIdx.x] = local;
  __syncthreads();
  for (int s = 128; s > 0; s >>= 1) {
    if (threadIdx.x < s) sred[threadIdx.x] += sred[threadIdx.x + s];
    __syncthreads();
  }
  if (threadIdx.x == 0) partials[blockIdx.x] = sred[0];
}

// ---------------- finalize: loss (fixed-order sum) + bincount -> out ----------------
__global__ void vq_finalize_kernel(const float* __restrict__ partials,
                                   const int* __restrict__ bin_ws,
                                   float* __restrict__ loss_out,
                                   float* __restrict__ bins_out) {
  int t = blockIdx.x * blockDim.x + threadIdx.x;
  if (t < K_CODE) bins_out[t] = (float)bin_ws[t];
  if (t == 0) {
    float s = 0.0f;
    for (int i = 0; i < 64; ++i) s += partials[i];  // deterministic order
    // loss = BETA*mse + mse = 1.25 * sum / (N*d)
    *loss_out = 1.25f * s / ((float)N_VEC * (float)DIM);
  }
}

extern "C" void kernel_launch(void* const* d_in, const int* in_sizes, int n_in,
                              void* d_out, int out_size, void* d_ws, size_t ws_size,
                              hipStream_t stream) {
  const float* z     = (const float*)d_in[0];   // [16,64,32,32]
  const float* embed = (const float*)d_in[1];   // [8192,64]

  // workspace layout (all offsets 32B-aligned for v16h loads)
  char* ws = (char*)d_ws;
  _Float16* zn       = (_Float16*)(ws);                                   // 16384*64*2 = 2 MiB
  _Float16* cn       = (_Float16*)(ws + (size_t)N_VEC * DIM * 2);         // 8192*64*2  = 1 MiB
  int*      idx_ws   = (int*)(ws + (size_t)N_VEC * DIM * 2
                                 + (size_t)K_CODE * DIM * 2);             // 64 KiB
  int*      bin_ws   = (int*)((char*)idx_ws + (size_t)N_VEC * 4);         // 32 KiB
  float*    partials = (float*)((char*)bin_ws + (size_t)K_CODE * 4);      // 256 B

  // output layout: [out 1048576][loss 1][indices 16384][bins 8192]
  float* out      = (float*)d_out;
  float* loss_out = out + (size_t)16 * 64 * 32 * 32;
  float* idx_out  = loss_out + 1;
  float* bins_out = idx_out + N_VEC;

  vq_init_kernel    <<<32,  256, 0, stream>>>(bin_ws, partials);
  vq_norm_cb_kernel <<<32,  256, 0, stream>>>(embed, cn);
  vq_norm_z_kernel  <<<64,  256, 0, stream>>>(z, zn);
  vq_argmax_kernel  <<<512, 256, 0, stream>>>(zn, cn, idx_ws);  // 4096 waves: 2 row-tiles x 4 K-chunks per block
  vq_gather_kernel  <<<64,  256, 0, stream>>>(z, embed, idx_ws, out, idx_out, bin_ws, partials);
  vq_finalize_kernel<<<32,  256, 0, stream>>>(partials, bin_ws, loss_out, bins_out);
}
